// SparseMoE_34411277975755
// MI455X (gfx1250) — matrix operand order repaired and checked by reading, not compiled, not run on hardware
//
#include <hip/hip_runtime.h>

#define TOK    4096
#define DMODEL 1024
#define DFFN   4096
#define NEXP   8
#define PAIRS  8192   // top_k * TOK

typedef __bf16        v16bf __attribute__((ext_vector_type(16)));
typedef float         v8f   __attribute__((ext_vector_type(8)));
typedef unsigned int  u32x4 __attribute__((ext_vector_type(4)));
typedef float         f32x4 __attribute__((ext_vector_type(4)));

__device__ __forceinline__ unsigned short f2bf(float f) {
  unsigned int u = __float_as_uint(f);
  u += 0x7FFFu + ((u >> 16) & 1u);          // round-to-nearest-even
  return (unsigned short)(u >> 16);
}

// CDNA5 async global->LDS copy (ASYNCcnt-tracked, no VGPR data path).
// vdst = per-lane LDS byte address, vaddr = per-lane 64-bit global address.
__device__ __forceinline__ void async_ld_b128(unsigned lds_off, const void* g) {
  unsigned long long ga = (unsigned long long)(uintptr_t)g;
  asm volatile("global_load_async_to_lds_b128 %0, %1, off"
               :: "v"(lds_off), "v"(ga) : "memory");
}
__device__ __forceinline__ void wait_async0() {
  asm volatile("s_wait_asynccnt 0x0" ::: "memory");
}

// ---------------- x fp32 -> bf16 (once) ----------------
__global__ void moe_cvt_x(const float* __restrict__ x, unsigned short* __restrict__ xb) {
  int i = blockIdx.x * blockDim.x + threadIdx.x;      // one float4 per thread
  f32x4 v = ((const f32x4*)x)[i];
  unsigned long long pk = (unsigned long long)f2bf(v[0])
                        | ((unsigned long long)f2bf(v[1]) << 16)
                        | ((unsigned long long)f2bf(v[2]) << 32)
                        | ((unsigned long long)f2bf(v[3]) << 48);
  ((unsigned long long*)xb)[i] = pk;
}

// ---------------- noisy top-2 router: one wave32 per token ----------------
__global__ void moe_router(const float* __restrict__ x, const float* __restrict__ noise,
                           const float* __restrict__ Wg, const float* __restrict__ bg,
                           const float* __restrict__ Wn, const float* __restrict__ bn,
                           int* __restrict__ tk_e, float* __restrict__ tk_g) {
  const int lane = threadIdx.x & 31;
  const int t = blockIdx.x * 8 + (threadIdx.x >> 5);
  float ag[NEXP] = {}, an[NEXP] = {};
  const float* xr = x + (size_t)t * DMODEL;
  for (int i = 0; i < DMODEL / 32; ++i) {
    int d = lane + 32 * i;
    float xv = xr[d];
    const f32x4* g = (const f32x4*)(Wg + d * NEXP);
    const f32x4* n = (const f32x4*)(Wn + d * NEXP);
    f32x4 g0 = g[0], g1 = g[1], n0 = n[0], n1 = n[1];
#pragma unroll
    for (int e = 0; e < 4; ++e) {
      ag[e]     += xv * g0[e];  ag[4 + e] += xv * g1[e];
      an[e]     += xv * n0[e];  an[4 + e] += xv * n1[e];
    }
  }
#pragma unroll
  for (int e = 0; e < NEXP; ++e)
    for (int off = 16; off; off >>= 1) {
      ag[e] += __shfl_xor(ag[e], off);
      an[e] += __shfl_xor(an[e], off);
    }
  if (lane == 0) {
    float nv[NEXP];
#pragma unroll
    for (int e = 0; e < NEXP; ++e) {
      float lg = ag[e] + bg[e];
      float nl = an[e] + bn[e];
      float sp = nl > 20.f ? nl : log1pf(expf(nl));      // softplus
      nv[e] = lg + noise[(size_t)t * NEXP + e] * sp;
    }
    int i1 = 0; float v1 = nv[0];
    for (int e = 1; e < NEXP; ++e) if (nv[e] > v1) { v1 = nv[e]; i1 = e; }
    int i2 = (i1 == 0) ? 1 : 0; float v2 = nv[i2];
    for (int e = 0; e < NEXP; ++e) if (e != i1 && nv[e] > v2) { v2 = nv[e]; i2 = e; }
    float e2v = expf(v2 - v1);                 // softmax over the two kept logits
    float gA = 1.f / (1.f + e2v);
    float gB = e2v / (1.f + e2v);
    tk_e[2 * t] = i1;  tk_e[2 * t + 1] = i2;
    tk_g[2 * t] = gA;  tk_g[2 * t + 1] = gB;
  }
}

// ---------------- deterministic count + exclusive scan ----------------
__global__ void moe_scan(const int* __restrict__ tk_e, int* __restrict__ cnt,
                         int* __restrict__ base) {
  if (blockIdx.x == 0 && threadIdx.x == 0) {
    int c[NEXP];
    for (int e = 0; e < NEXP; ++e) c[e] = 0;
    for (int p = 0; p < PAIRS; ++p) c[tk_e[p]]++;
    int s = 0;
    for (int e = 0; e < NEXP; ++e) { cnt[e] = c[e]; base[e] = s; s += c[e]; }
  }
}

// ---------------- deterministic per-expert compaction ----------------
__global__ void moe_fill(const int* __restrict__ tk_e, const int* __restrict__ base,
                         int* __restrict__ slot_token, int* __restrict__ pair_slot) {
  int e = blockIdx.x;
  if (threadIdx.x != 0) return;
  int pos = base[e];
  for (int p = 0; p < PAIRS; ++p)
    if (tk_e[p] == e) { slot_token[pos] = p >> 1; pair_slot[p] = pos; ++pos; }
}

// ======================================================================
// Double-buffered grouped-GEMM bodies.
//   A tile (bf16, no transform):  async global->LDS (ASYNCcnt)
//   B tile (fp32 -> bf16, transposed): global->regs, pack pairs, b32 LDS store
// Pipeline per K-step: issue(A_next async, B_next fetch) -> WMMA(cur)
//                      -> commit B_next -> s_wait_asynccnt 0 -> barrier
// ======================================================================
#define LDSW 40   // halfwords per LDS row: 80B, 16B-aligned, padded

__global__ __launch_bounds__(256)
void moe_fc1(const unsigned short* __restrict__ xb,
             const float* __restrict__ W1, const float* __restrict__ b1,
             const int* __restrict__ cnt, const int* __restrict__ base,
             const int* __restrict__ slot_token, unsigned short* __restrict__ H) {
  __shared__ unsigned short lA[2][128 * LDSW];
  __shared__ unsigned short lB[2][128 * LDSW];

  const int e = blockIdx.z;
  const int c = cnt[e];
  const int mbase = blockIdx.y * 128;
  if (mbase >= c) return;
  const int sbase = base[e];
  const int nbase = blockIdx.x * 128;
  const float* W = W1 + (size_t)e * DMODEL * DFFN;

  const int tid = threadIdx.x, lane = tid & 31, wave = tid >> 5;
  const int wm = wave & 3, wn = wave >> 2;

  // ---- A geometry: 2 x 16B chunks per thread, gathered token rows ----
  const unsigned short* srcA[2];
  unsigned aoff[2][2];
#pragma unroll
  for (int i = 0; i < 2; ++i) {
    int chunk = tid + 256 * i;
    int row = chunk >> 2, part = chunk & 3;
    int r = mbase + row;  if (r >= c) r = c - 1;           // clamp padded rows
    srcA[i] = xb + (size_t)slot_token[sbase + r] * DMODEL + part * 8;
    aoff[0][i] = (unsigned)(uintptr_t)&lA[0][row * LDSW + part * 8];
    aoff[1][i] = (unsigned)(uintptr_t)&lA[1][row * LDSW + part * 8];
  }
  // ---- B geometry: 2 iters x (2 K-rows x 4 F-cols) per thread ----
  int kk2[2], f4[2];
#pragma unroll
  for (int i = 0; i < 2; ++i) {
    int q = tid + 256 * i;
    kk2[i] = q >> 5;  f4[i] = (q & 31) * 4;
  }
  f32x4 wreg[2][2];

  auto asyncA = [&](int k0, int buf) {
#pragma unroll
    for (int i = 0; i < 2; ++i) async_ld_b128(aoff[buf][i], srcA[i] + k0);
  };
  auto fetchB = [&](int k0) {
#pragma unroll
    for (int i = 0; i < 2; ++i) {
      const float* wp = W + (size_t)(k0 + 2 * kk2[i]) * DFFN + nbase + f4[i];
      wreg[i][0] = *(const f32x4*)wp;
      wreg[i][1] = *(const f32x4*)(wp + DFFN);
      if (k0 + 32 < DMODEL) __builtin_prefetch(wp + (size_t)32 * DFFN, 0, 0);
    }
  };
  auto commitB = [&](int buf) {
    unsigned short* lb = lB[buf];
#pragma unroll
    for (int i = 0; i < 2; ++i)
#pragma unroll
      for (int j = 0; j < 4; ++j) {
        unsigned pk = (unsigned)f2bf(wreg[i][0][j])
                    | ((unsigned)f2bf(wreg[i][1][j]) << 16);
        *(unsigned*)(lb + (f4[i] + j) * LDSW + 2 * kk2[i]) = pk;
      }
  };

  v8f acc[2][4] = {};
  const int half = lane >> 4, l16 = lane & 15;

  asyncA(0, 0);
  fetchB(0);
  commitB(0);
  wait_async0();
  __syncthreads();

  const int NK = DMODEL / 32;
  for (int kt = 0; kt < NK; ++kt) {
    const int cur = kt & 1;
    const bool more = (kt + 1 < NK);
    if (more) { asyncA((kt + 1) * 32, cur ^ 1); fetchB((kt + 1) * 32); }

    union Frag { v16bf v; u32x4 q[2]; } a[2], b[4];
#pragma unroll
    for (int m = 0; m < 2; ++m) {              // A: K 0-7|16-23 vs 8-15|24-31 per half
      const unsigned short* p = lA[cur] + (wm * 32 + m * 16 + l16) * LDSW + half * 8;
      a[m].q[0] = *(const u32x4*)p;
      a[m].q[1] = *(const u32x4*)(p + 16);
    }
#pragma unroll
    for (int n = 0; n < 4; ++n) {              // B: K 0-7|16-23 vs 8-15|24-31 per half
      const unsigned short* p = lB[cur] + (wn * 64 + n * 16 + l16) * LDSW + half * 8;
      b[n].q[0] = *(const u32x4*)p;
      b[n].q[1] = *(const u32x4*)(p + 16);
    }
#pragma unroll
    for (int m = 0; m < 2; ++m)
#pragma unroll
      for (int n = 0; n < 4; ++n)
        acc[m][n] = __builtin_amdgcn_wmma_f32_16x16x32_bf16(
            false, a[m].v, false, b[n].v, (short)0, acc[m][n], false, false);

    if (more) { commitB(cur ^ 1); wait_async0(); }
    __syncthreads();
  }

#pragma unroll
  for (int n = 0; n < 4; ++n) {
    int col = nbase + wn * 64 + n * 16 + l16;
    float bb = b1[e * DFFN + col];
#pragma unroll
    for (int m = 0; m < 2; ++m)
#pragma unroll
      for (int r = 0; r < 8; ++r) {
        int row = mbase + wm * 32 + m * 16 + r + 8 * half;
        if (row < c) {
          float v = acc[m][n][r] + bb;
          H[(size_t)(sbase + row) * DFFN + col] = f2bf(v > 0.f ? v : 0.f);
        }
      }
  }
}

__global__ __launch_bounds__(256)
void moe_fc2(const unsigned short* __restrict__ H,
             const float* __restrict__ W2, const float* __restrict__ b2,
             const int* __restrict__ cnt, const int* __restrict__ base,
             float* __restrict__ EO) {
  __shared__ unsigned short lA[2][128 * LDSW];
  __shared__ unsigned short lB[2][128 * LDSW];

  const int e = blockIdx.z;
  const int c = cnt[e];
  const int mbase = blockIdx.y * 128;
  if (mbase >= c) return;
  const int sbase = base[e];
  const int nbase = blockIdx.x * 128;
  const float* W = W2 + (size_t)e * DFFN * DMODEL;

  const int tid = threadIdx.x, lane = tid & 31, wave = tid >> 5;
  const int wm = wave & 3, wn = wave >> 2;

  const unsigned short* srcA[2];
  unsigned aoff[2][2];
#pragma unroll
  for (int i = 0; i < 2; ++i) {
    int chunk = tid + 256 * i;
    int row = chunk >> 2, part = chunk & 3;
    int r = mbase + row;  if (r >= c) r = c - 1;
    srcA[i] = H + (size_t)(sbase + r) * DFFN + part * 8;
    aoff[0][i] = (unsigned)(uintptr_t)&lA[0][row * LDSW + part * 8];
    aoff[1][i] = (unsigned)(uintptr_t)&lA[1][row * LDSW + part * 8];
  }
  int kk2[2], f4[2];
#pragma unroll
  for (int i = 0; i < 2; ++i) {
    int q = tid + 256 * i;
    kk2[i] = q >> 5;  f4[i] = (q & 31) * 4;
  }
  f32x4 wreg[2][2];

  auto asyncA = [&](int k0, int buf) {
#pragma unroll
    for (int i = 0; i < 2; ++i) async_ld_b128(aoff[buf][i], srcA[i] + k0);
  };
  auto fetchB = [&](int k0) {
#pragma unroll
    for (int i = 0; i < 2; ++i) {
      const float* wp = W + (size_t)(k0 + 2 * kk2[i]) * DMODEL + nbase + f4[i];
      wreg[i][0] = *(const f32x4*)wp;
      wreg[i][1] = *(const f32x4*)(wp + DMODEL);
      if (k0 + 32 < DFFN) __builtin_prefetch(wp + (size_t)32 * DMODEL, 0, 0);
    }
  };
  auto commitB = [&](int buf) {
    unsigned short* lb = lB[buf];
#pragma unroll
    for (int i = 0; i < 2; ++i)
#pragma unroll
      for (int j = 0; j < 4; ++j) {
        unsigned pk = (unsigned)f2bf(wreg[i][0][j])
                    | ((unsigned)f2bf(wreg[i][1][j]) << 16);
        *(unsigned*)(lb + (f4[i] + j) * LDSW + 2 * kk2[i]) = pk;
      }
  };

  v8f acc[2][4] = {};
  const int half = lane >> 4, l16 = lane & 15;

  asyncA(0, 0);
  fetchB(0);
  commitB(0);
  wait_async0();
  __syncthreads();

  const int NK = DFFN / 32;
  for (int kt = 0; kt < NK; ++kt) {
    const int cur = kt & 1;
    const bool more = (kt + 1 < NK);
    if (more) { asyncA((kt + 1) * 32, cur ^ 1); fetchB((kt + 1) * 32); }

    union Frag { v16bf v; u32x4 q[2]; } a[2], b[4];
#pragma unroll
    for (int m = 0; m < 2; ++m) {
      const unsigned short* p = lA[cur] + (wm * 32 + m * 16 + l16) * LDSW + half * 8;
      a[m].q[0] = *(const u32x4*)p;
      a[m].q[1] = *(const u32x4*)(p + 16);
    }
#pragma unroll
    for (int n = 0; n < 4; ++n) {
      const unsigned short* p = lB[cur] + (wn * 64 + n * 16 + l16) * LDSW + half * 8;
      b[n].q[0] = *(const u32x4*)p;
      b[n].q[1] = *(const u32x4*)(p + 16);
    }
#pragma unroll
    for (int m = 0; m < 2; ++m)
#pragma unroll
      for (int n = 0; n < 4; ++n)
        acc[m][n] = __builtin_amdgcn_wmma_f32_16x16x32_bf16(
            false, a[m].v, false, b[n].v, (short)0, acc[m][n], false, false);

    if (more) { commitB(cur ^ 1); wait_async0(); }
    __syncthreads();
  }

#pragma unroll
  for (int n = 0; n < 4; ++n) {
    int col = nbase + wn * 64 + n * 16 + l16;
    float bb = b2[e * DMODEL + col];
#pragma unroll
    for (int m = 0; m < 2; ++m)
#pragma unroll
      for (int r = 0; r < 8; ++r) {
        int row = mbase + wm * 32 + m * 16 + r + 8 * half;
        if (row < c)
          EO[(size_t)(sbase + row) * DMODEL + col] = acc[m][n][r] + bb;
      }
  }
}

// ---------------- combine: out[t] = g0*EO[slot0] + g1*EO[slot1] ----------------
__global__ void moe_combine(const float* __restrict__ EO, const int* __restrict__ pair_slot,
                            const float* __restrict__ tk_g, float* __restrict__ out) {
  int i = blockIdx.x * blockDim.x + threadIdx.x;   // TOK*DMODEL/4 threads
  int t = i >> 8;                                   // 256 float4 per token
  int d4 = (i & 255) * 4;
  int s0 = pair_slot[2 * t], s1 = pair_slot[2 * t + 1];
  float g0 = tk_g[2 * t], g1 = tk_g[2 * t + 1];
  f32x4 a = *(const f32x4*)(EO + (size_t)s0 * DMODEL + d4);
  f32x4 b = *(const f32x4*)(EO + (size_t)s1 * DMODEL + d4);
  f32x4 r = { a[0] * g0 + b[0] * g1, a[1] * g0 + b[1] * g1,
              a[2] * g0 + b[2] * g1, a[3] * g0 + b[3] * g1 };
  *(f32x4*)(out + (size_t)t * DMODEL + d4) = r;
}

extern "C" void kernel_launch(void* const* d_in, const int* in_sizes, int n_in,
                              void* d_out, int out_size, void* d_ws, size_t ws_size,
                              hipStream_t stream) {
  (void)in_sizes; (void)n_in; (void)out_size; (void)ws_size;
  const float* x     = (const float*)d_in[0];
  const float* noise = (const float*)d_in[1];
  const float* Wg    = (const float*)d_in[2];
  const float* bg    = (const float*)d_in[3];
  const float* Wn    = (const float*)d_in[4];
  const float* bn    = (const float*)d_in[5];
  const float* W1    = (const float*)d_in[6];
  const float* b1    = (const float*)d_in[7];
  const float* W2    = (const float*)d_in[8];
  const float* b2    = (const float*)d_in[9];
  float* out = (float*)d_out;

  // workspace layout (~104 MiB)
  char* ws = (char*)d_ws;
  constexpr size_t O_H    = 0;
  constexpr size_t O_EO   = O_H  + (size_t)PAIRS * DFFN * 2;   // H  bf16: 64 MiB
  constexpr size_t O_XB   = O_EO + (size_t)PAIRS * DMODEL * 4; // EO fp32: 32 MiB
  constexpr size_t O_ST   = O_XB + (size_t)TOK * DMODEL * 2;   // xb bf16:  8 MiB
  constexpr size_t O_TE   = O_ST + (size_t)PAIRS * 4;
  constexpr size_t O_TG   = O_TE + (size_t)PAIRS * 4;
  constexpr size_t O_PS   = O_TG + (size_t)PAIRS * 4;
  constexpr size_t O_CNT  = O_PS + (size_t)PAIRS * 4;
  constexpr size_t O_BASE = O_CNT + 64;

  unsigned short* H  = (unsigned short*)(ws + O_H);
  float* EO          = (float*)(ws + O_EO);
  unsigned short* xb = (unsigned short*)(ws + O_XB);
  int*   slot_token  = (int*)(ws + O_ST);
  int*   tk_e        = (int*)(ws + O_TE);
  float* tk_g        = (float*)(ws + O_TG);
  int*   pair_slot   = (int*)(ws + O_PS);
  int*   cnt         = (int*)(ws + O_CNT);
  int*   base        = (int*)(ws + O_BASE);

  moe_cvt_x <<<TOK * DMODEL / 4 / 256, 256, 0, stream>>>(x, xb);
  moe_router<<<TOK / 8, 256, 0, stream>>>(x, noise, Wg, bg, Wn, bn, tk_e, tk_g);
  moe_scan  <<<1, 1, 0, stream>>>(tk_e, cnt, base);
  moe_fill  <<<NEXP, 1, 0, stream>>>(tk_e, base, slot_token, pair_slot);
  moe_fc1   <<<dim3(DFFN / 128, PAIRS / 128, NEXP), 256, 0, stream>>>(
      xb, W1, b1, cnt, base, slot_token, H);
  moe_fc2   <<<dim3(DMODEL / 128, PAIRS / 128, NEXP), 256, 0, stream>>>(
      H, W2, b2, cnt, base, EO);
  moe_combine<<<TOK * DMODEL / 4 / 256, 256, 0, stream>>>(EO, pair_slot, tk_g, out);
}
